// GatedGNN_87333864997007
// MI455X (gfx1250) — compile-verified, matrix operand order checked by reading, not run on hardware
//
#include <hip/hip_runtime.h>

// ---------------- problem constants (from reference) ----------------
#define Nn    100000      // nodes (divisible by 16 -> 6250 row tiles)
#define Ee    3200000     // edges
#define Hh    128         // hidden = F_IN
#define Cc    40          // classes
#define CPAD  48          // padded to 3 WMMA col tiles
#define BNEPS 1e-5f

typedef __attribute__((ext_vector_type(16))) __bf16 v16bf;
typedef __attribute__((ext_vector_type(8)))  float  v8f;

// ---------------- WMMA helpers (gfx1250, wave32) ----------------
__device__ __forceinline__ __bf16 f2bf(float f) { return (__bf16)f; }

__device__ __forceinline__ v8f wmma_bf16(v16bf a, v16bf b, v8f c) {
  // D = A(16x32 bf16) * B(32x16 bf16) + C(16x16 f32)
  return __builtin_amdgcn_wmma_f32_16x16x32_bf16(false, a, false, b, (short)0, c,
                                                 false, false);
}

// A fragment: 16x32 tile, rows at `base` (row-major, stride ld floats), k offset k0.
// ISA 7.12.2 16-bit A layout: lanes 0-15 -> M=0..15 (K 0..7,16..23),
// lanes 16-31 -> same M, K 8..15,24..31; two bf16 per VGPR.
__device__ __forceinline__ v16bf load_a(const float* __restrict__ base, int ld, int k0) {
  int lane = threadIdx.x & 31;
  int m    = lane & 15;
  int kb   = (lane >> 4) << 3;             // 0 or 8
  const float* p = base + (size_t)m * ld + k0;
  v16bf a;
#pragma unroll
  for (int e = 0; e < 16; ++e) {
    int ko = (e < 8) ? (kb + e) : (16 + kb + (e - 8));
    a[e] = f2bf(p[ko]);
  }
  return a;
}

// B fragment from pre-packed buffer: tile-major, then lane-major, 16 bf16 each.
__device__ __forceinline__ v16bf load_b(const __bf16* __restrict__ pk, int tile) {
  int lane = threadIdx.x & 31;
  return *(const v16bf*)(pk + ((size_t)tile * 32 + lane) * 16);  // 32B aligned
}

__device__ __forceinline__ float sigmoidf_(float x) { return 1.f / (1.f + __expf(-x)); }

// ---------------- weight packing into B-fragment order ----------------
// B logical layout is K x Ncols (Ncols padded to 16). transposed=1 means the
// fp32 source is [Ncols, K] row-major (for GRU W^T GEMMs).
__global__ void k_pack(const float* __restrict__ src, __bf16* __restrict__ dst,
                       int K, int Ncols, int realN, int transposed, int total) {
  int idx = blockIdx.x * blockDim.x + threadIdx.x;
  if (idx >= total) return;
  int e    = idx & 15;
  int lane = (idx >> 4) & 31;
  int tile = idx >> 9;
  int NT   = Ncols >> 4;
  int nt   = tile % NT;
  int kc   = tile / NT;
  int n    = nt * 16 + (lane & 15);
  int kb   = (lane >> 4) << 3;
  int ko   = (e < 8) ? (kb + e) : (16 + kb + (e - 8));
  int k    = kc * 32 + ko;
  float v  = 0.f;
  if (n < realN) v = transposed ? src[(size_t)n * K + k] : src[(size_t)k * realN + n];
  dst[idx] = f2bf(v);
}

// ---------------- zero fill ----------------
__global__ void k_zero(float* __restrict__ p, long long n) {
  long long i = (long long)blockIdx.x * blockDim.x + threadIdx.x;
  long long s = (long long)gridDim.x * blockDim.x;
  for (; i < n; i += s) p[i] = 0.f;
}

// ---------------- embedding: x = h @ embW + b ; hin = x ----------------
__global__ void k_embed(const float* __restrict__ h, const __bf16* __restrict__ Bpk,
                        const float* __restrict__ bias, float* __restrict__ x,
                        float* __restrict__ hin) {
  int tile = blockIdx.x * 4 + (threadIdx.x >> 5);
  if (tile >= Nn / 16) return;                      // wave-uniform exit
  int lane = threadIdx.x & 31;
  int col = lane & 15, rh = (lane >> 4) << 3;
  const float* hb = h + (size_t)tile * 16 * Hh;
  v16bf aF[4];
#pragma unroll
  for (int kc = 0; kc < 4; ++kc) aF[kc] = load_a(hb, Hh, kc * 32);
#pragma unroll
  for (int j = 0; j < 8; ++j) {
    v8f acc;
    float bv = bias[j * 16 + col];
#pragma unroll
    for (int i = 0; i < 8; ++i) acc[i] = bv;
#pragma unroll
    for (int kc = 0; kc < 4; ++kc) acc = wmma_bf16(aF[kc], load_b(Bpk, kc * 8 + j), acc);
#pragma unroll
    for (int i = 0; i < 8; ++i) {
      size_t off = (size_t)(tile * 16 + i + rh) * Hh + j * 16 + col;
      x[off] = acc[i];
      hin[off] = acc[i];
    }
  }
}

// ---------------- conv: m = x @ Wc ----------------
__global__ void k_conv(const float* __restrict__ x, const __bf16* __restrict__ Bpk,
                       float* __restrict__ m) {
  int tile = blockIdx.x * 4 + (threadIdx.x >> 5);
  if (tile >= Nn / 16) return;
  int lane = threadIdx.x & 31;
  int col = lane & 15, rh = (lane >> 4) << 3;
  const float* xb = x + (size_t)tile * 16 * Hh;
  v16bf aF[4];
#pragma unroll
  for (int kc = 0; kc < 4; ++kc) aF[kc] = load_a(xb, Hh, kc * 32);
#pragma unroll
  for (int j = 0; j < 8; ++j) {
    v8f acc = {};
#pragma unroll
    for (int kc = 0; kc < 4; ++kc) acc = wmma_bf16(aF[kc], load_b(Bpk, kc * 8 + j), acc);
#pragma unroll
    for (int i = 0; i < 8; ++i)
      m[(size_t)(tile * 16 + i + rh) * Hh + j * 16 + col] = acc[i];
  }
}

// ---------------- edge gather + weighted scatter-add (L2-resident) ----------
// one wave per edge, lane -> 4 features; agg/m are 51MB each -> live in 192MB L2
__global__ void k_scatter(const float* __restrict__ m, const long long* __restrict__ ei,
                          const float* __restrict__ ew, float* __restrict__ agg) {
  long long e = (long long)blockIdx.x * 8 + (threadIdx.x >> 5);
  if (e >= Ee) return;
  int lane = threadIdx.x & 31;
  long long s = ei[e];
  long long d = ei[(long long)Ee + e];
  float w = ew[e];
  const float4 v = *(const float4*)(m + (size_t)s * Hh + lane * 4);
  float* dp = agg + (size_t)d * Hh + lane * 4;
  unsafeAtomicAdd(dp + 0, w * v.x);
  unsafeAtomicAdd(dp + 1, w * v.y);
  unsafeAtomicAdd(dp + 2, w * v.z);
  unsafeAtomicAdd(dp + 3, w * v.w);
}

// ---------------- fused GRU cell: gi = agg@Wih^T+bih ; gh = x@Whh^T+bhh -----
__global__ void k_gru(float* __restrict__ x, const float* __restrict__ agg,
                      const __bf16* __restrict__ WihPk, const __bf16* __restrict__ WhhPk,
                      const float* __restrict__ bih, const float* __restrict__ bhh) {
  int tile = blockIdx.x * 4 + (threadIdx.x >> 5);
  if (tile >= Nn / 16) return;
  int lane = threadIdx.x & 31;
  int col = lane & 15, rh = (lane >> 4) << 3;
  const float* ab = agg + (size_t)tile * 16 * Hh;
  float* xb = x + (size_t)tile * 16 * Hh;
  v16bf aA[4], aX[4];
#pragma unroll
  for (int kc = 0; kc < 4; ++kc) {
    aA[kc] = load_a(ab, Hh, kc * 32);
    aX[kc] = load_a(xb, Hh, kc * 32);
  }
#pragma unroll
  for (int j = 0; j < 8; ++j) {
    v8f ir, hr, iz, hz, in_, hn;
    float bir = bih[0 * Hh + j * 16 + col], bhr = bhh[0 * Hh + j * 16 + col];
    float biz = bih[1 * Hh + j * 16 + col], bhz = bhh[1 * Hh + j * 16 + col];
    float bin = bih[2 * Hh + j * 16 + col], bhn = bhh[2 * Hh + j * 16 + col];
#pragma unroll
    for (int i = 0; i < 8; ++i) {
      ir[i] = bir; hr[i] = bhr; iz[i] = biz; hz[i] = bhz; in_[i] = bin; hn[i] = bhn;
    }
#pragma unroll
    for (int kc = 0; kc < 4; ++kc) {               // 24 B-tiles per K-chunk (3H=384)
      ir  = wmma_bf16(aA[kc], load_b(WihPk, kc * 24 + 0 * 8 + j), ir);
      hr  = wmma_bf16(aX[kc], load_b(WhhPk, kc * 24 + 0 * 8 + j), hr);
      iz  = wmma_bf16(aA[kc], load_b(WihPk, kc * 24 + 1 * 8 + j), iz);
      hz  = wmma_bf16(aX[kc], load_b(WhhPk, kc * 24 + 1 * 8 + j), hz);
      in_ = wmma_bf16(aA[kc], load_b(WihPk, kc * 24 + 2 * 8 + j), in_);
      hn  = wmma_bf16(aX[kc], load_b(WhhPk, kc * 24 + 2 * 8 + j), hn);
    }
#pragma unroll
    for (int i = 0; i < 8; ++i) {                  // gate math on C/D layout
      size_t off = (size_t)(i + rh) * Hh + j * 16 + col;
      float xo = xb[off];
      float r = sigmoidf_(ir[i] + hr[i]);
      float z = sigmoidf_(iz[i] + hz[i]);
      float n = tanhf(in_[i] + r * hn[i]);
      xb[off] = (1.f - z) * n + z * xo;            // in place: rows/cols wave-private
    }
  }
}

// ---------------- BatchNorm statistics ----------------
__global__ void k_bnstats(const float* __restrict__ x, float* __restrict__ sums) {
  int g = blockIdx.x * blockDim.x + threadIdx.x;   // 128 blocks x 256
  int f = g & 127;
  int r0 = g >> 7;
  int stride = (gridDim.x * blockDim.x) >> 7;
  float s = 0.f, q = 0.f;
  for (int r = r0; r < Nn; r += stride) {
    float v = x[(size_t)r * Hh + f];
    s += v; q += v * v;
  }
  unsafeAtomicAdd(&sums[f], s);
  unsafeAtomicAdd(&sums[Hh + f], q);
}

__global__ void k_bnparams(const float* __restrict__ sums, const float* __restrict__ gamma,
                           const float* __restrict__ beta, float* __restrict__ sT) {
  int f = threadIdx.x;
  if (f < Hh) {
    float invN = 1.f / (float)Nn;
    float mean = sums[f] * invN;
    float var  = sums[Hh + f] * invN - mean * mean;
    float s    = gamma[f] * rsqrtf(var + BNEPS);
    sT[f]      = s;
    sT[Hh + f] = beta[f] - mean * s;               // y = x*s + t
  }
}

// ---------------- BN + residual + classifier GEMM ----------------
__global__ void k_final(const float* __restrict__ x, const float* __restrict__ hin,
                        const float* __restrict__ sT, const __bf16* __restrict__ mlpPk,
                        const float* __restrict__ mlpb, float* __restrict__ out) {
  int tile = blockIdx.x * 4 + (threadIdx.x >> 5);
  if (tile >= Nn / 16) return;
  int lane = threadIdx.x & 31;
  int m_ = lane & 15, kb = (lane >> 4) << 3;
  int col = lane & 15, rh = (lane >> 4) << 3;
  const float* xb = x + (size_t)tile * 16 * Hh;
  const float* hb = hin + (size_t)tile * 16 * Hh;
  v16bf aF[4];
#pragma unroll
  for (int kc = 0; kc < 4; ++kc) {
    const float* px = xb + (size_t)m_ * Hh + kc * 32;
    const float* ph = hb + (size_t)m_ * Hh + kc * 32;
    v16bf a;
#pragma unroll
    for (int e = 0; e < 16; ++e) {
      int ko = (e < 8) ? (kb + e) : (16 + kb + (e - 8));
      int k = kc * 32 + ko;
      a[e] = f2bf(px[ko] * sT[k] + sT[Hh + k] + ph[ko]);   // norm + residual
    }
    aF[kc] = a;
  }
#pragma unroll
  for (int j = 0; j < 3; ++j) {                    // 48 padded cols = 3 tiles
    int c = j * 16 + col;
    float bv = (c < Cc) ? mlpb[c] : 0.f;
    v8f acc;
#pragma unroll
    for (int i = 0; i < 8; ++i) acc[i] = bv;
#pragma unroll
    for (int kc = 0; kc < 4; ++kc) acc = wmma_bf16(aF[kc], load_b(mlpPk, kc * 3 + j), acc);
    if (c < Cc) {
#pragma unroll
      for (int i = 0; i < 8; ++i)
        out[(size_t)(tile * 16 + i + rh) * Cc + c] = acc[i];
    }
  }
}

// ---------------- host launch ----------------
extern "C" void kernel_launch(void* const* d_in, const int* in_sizes, int n_in,
                              void* d_out, int out_size, void* d_ws, size_t ws_size,
                              hipStream_t stream) {
  const float*     h     = (const float*)d_in[0];
  const long long* ei    = (const long long*)d_in[1];   // jnp.int64 edge_index
  const float*     ew    = (const float*)d_in[2];
  const float*     embW  = (const float*)d_in[3];
  const float*     embB  = (const float*)d_in[4];
  const float*     convW = (const float*)d_in[5];
  const float*     Wih   = (const float*)d_in[6];
  const float*     Whh   = (const float*)d_in[7];
  const float*     bih   = (const float*)d_in[8];
  const float*     bhh   = (const float*)d_in[9];
  const float*     gamma = (const float*)d_in[10];
  const float*     beta  = (const float*)d_in[11];
  const float*     mlpW  = (const float*)d_in[12];
  const float*     mlpb  = (const float*)d_in[13];
  float*           out   = (float*)d_out;

  // workspace carve-out (~205 MB): x, hin, m, agg in fp32 + packed bf16 weights
  char* ws = (char*)d_ws;
  size_t o = 0;
  auto carve = [&](size_t bytes) -> char* {
    char* p = ws + o;
    o = (o + bytes + 255) & ~(size_t)255;
    return p;
  };
  float*  x      = (float*)carve((size_t)Nn * Hh * 4);
  float*  hin    = (float*)carve((size_t)Nn * Hh * 4);
  float*  m      = (float*)carve((size_t)Nn * Hh * 4);
  float*  agg    = (float*)carve((size_t)Nn * Hh * 4);
  float*  sums   = (float*)carve(256 * 4);
  float*  sT     = (float*)carve(256 * 4);
  __bf16* embPk  = (__bf16*)carve(16384 * 2);     // 4 kc * 8 nt tiles
  __bf16* convPk = (__bf16*)carve(3 * 16384 * 2);
  __bf16* WihPk  = (__bf16*)carve(49152 * 2);     // 4 kc * 24 nt tiles
  __bf16* WhhPk  = (__bf16*)carve(49152 * 2);
  __bf16* mlpPk  = (__bf16*)carve(6144 * 2);      // 4 kc * 3 nt tiles

  // pack weights into WMMA B-fragment order (bf16)
  k_pack<<<(16384 + 255) / 256, 256, 0, stream>>>(embW, embPk, 128, 128, 128, 0, 16384);
  for (int i = 0; i < 3; ++i)
    k_pack<<<(16384 + 255) / 256, 256, 0, stream>>>(convW + (size_t)i * 128 * 128,
                                                    convPk + (size_t)i * 16384,
                                                    128, 128, 128, 0, 16384);
  k_pack<<<(49152 + 255) / 256, 256, 0, stream>>>(Wih, WihPk, 128, 384, 384, 1, 49152);
  k_pack<<<(49152 + 255) / 256, 256, 0, stream>>>(Whh, WhhPk, 128, 384, 384, 1, 49152);
  k_pack<<<(6144 + 255) / 256, 256, 0, stream>>>(mlpW, mlpPk, 128, CPAD, Cc, 0, 6144);

  const int gemmBlocks = (Nn / 16 + 3) / 4;       // 4 waves (16-row tiles) per block

  k_embed<<<gemmBlocks, 128, 0, stream>>>(h, embPk, embB, x, hin);

  for (int l = 0; l < 3; ++l) {
    k_zero<<<2048, 256, 0, stream>>>(agg, (long long)Nn * Hh);
    k_conv<<<gemmBlocks, 128, 0, stream>>>(x, convPk + (size_t)l * 16384, m);
    k_scatter<<<(Ee + 7) / 8, 256, 0, stream>>>(m, ei, ew, agg);
    k_gru<<<gemmBlocks, 128, 0, stream>>>(x, agg, WihPk, WhhPk, bih, bhh);
  }

  k_zero<<<1, 256, 0, stream>>>(sums, 256);
  k_bnstats<<<128, 256, 0, stream>>>(x, sums);
  k_bnparams<<<1, 128, 0, stream>>>(sums, gamma, beta, sT);
  k_final<<<gemmBlocks, 128, 0, stream>>>(x, hin, sT, mlpPk, mlpb, out);
}
